// BayesianFilter_32959579029620
// MI455X (gfx1250) — matrix operand order repaired and checked by reading, not compiled, not run on hardware
//
#include <hip/hip_runtime.h>
#include <hip/hip_bf16.h>

typedef __attribute__((ext_vector_type(16))) _Float16 v16h;
typedef __attribute__((ext_vector_type(8)))  float    v8f;

#define TPB 256
#define WPB 8   // waves per block (wave32)

__device__ __forceinline__ float act_apply(float x, int act) {
  if (act == 1) return x > 0.0f ? x : (__expf(x) - 1.0f);  // ELU
  if (act == 2) return tanhf(x);                            // tanh
  return x;
}

// ---------------------------------------------------------------------------
// LDS-tiled, double-buffered, software-pipelined batched GEMM on WMMA:
//   C[b] = act(beta*C[b] + A[b](MxK) @ B[b](KxN) + bias[n])
// Requirements (hold for every call site here): M%64==0, N%64==0, K%32==0.
// Block = 256 threads = 8 waves (2x4); block tile 64x64, k-step 32.
// Pipeline: barrier -> issue next tile's global_load_b128 (registers) ->
// fragments from current LDS buffer -> 2x WMMA -> cvt+ds_store next buffer.
// One barrier per k-step; global loads overlap the matrix ops.
// ---------------------------------------------------------------------------
__global__ void gemm_nn_wmma(const float* __restrict__ A, const float* __restrict__ B,
                             const float* __restrict__ bias, float* __restrict__ C,
                             int M, int N, int K, int nbatch,
                             long sA, long sB, long sC, int beta, int act)
{
  __shared__ _Float16 As[2][64 * 34];   // [row 0..63][k 0..31], stride 34
  __shared__ _Float16 Bs[2][64 * 34];   // transposed: [n 0..63][k 0..31], stride 34

  int tid  = threadIdx.x;
  int lane = tid & 31;
  int wv   = tid >> 5;
  int mt = M >> 6, nt = N >> 6;
  long tpb  = (long)mt * nt;
  long bidx = (long)blockIdx.x / tpb;
  if (bidx >= (long)nbatch) return;               // block-uniform (grid exact)
  long rr = (long)blockIdx.x - bidx * tpb;
  int tm = (int)(rr / nt), tn = (int)(rr % nt);
  const float* Ab = A + bidx * sA;
  const float* Bb = B + bidx * sB;
  float*       Cb = C + bidx * sC;
  int mBase = tm << 6, nBase = tn << 6;
  int wr = wv >> 2, wc = wv & 3;                  // 2x4 wave grid
  int ml = lane & 15, hi = lane >> 4;
  int nc = nBase + (wc << 4) + ml;                // this lane's output column

  // loader indices: 2 float4 for A + 2 for B per thread per 64x32 tile
  int arow[2], acol[2], brow[2], bcol[2];
#pragma unroll
  for (int it = 0; it < 2; ++it) {
    int idx = it * 256 + tid;
    arow[it] = idx >> 3;  acol[it] = (idx & 7) << 2;
    brow[it] = idx >> 4;  bcol[it] = (idx & 15) << 2;
  }

  v8f acc0 = {0.f, 0.f, 0.f, 0.f, 0.f, 0.f, 0.f, 0.f};
  v8f acc1 = acc0;
  int mw = mBase + (wr << 5);                     // wave's 32-row band
  if (beta) {
#pragma unroll
    for (int q = 0; q < 8; ++q) {
      acc0[q] = Cb[(long)(mw + q + 8 * hi) * N + nc];
      acc1[q] = Cb[(long)(mw + 16 + q + 8 * hi) * N + nc];
    }
  }

  float4 va[2], vb[2];
  auto fetchT = [&](int kk) {                     // global -> registers (b128 loads)
#pragma unroll
    for (int it = 0; it < 2; ++it) {
      va[it] = *(const float4*)&Ab[(long)(mBase + arow[it]) * K + kk + acol[it]];
      vb[it] = *(const float4*)&Bb[(long)(kk + brow[it]) * N + nBase + bcol[it]];
    }
  };
  auto stageT = [&](int bufI) {                   // registers -> f16 LDS tiles
    _Float16* Ad = As[bufI];
    _Float16* Bd = Bs[bufI];
#pragma unroll
    for (int it = 0; it < 2; ++it) {
      _Float16 t0[2] = {(_Float16)va[it].x, (_Float16)va[it].y};
      _Float16 t1[2] = {(_Float16)va[it].z, (_Float16)va[it].w};
      _Float16* p = &Ad[arow[it] * 34 + acol[it]];
      __builtin_memcpy(p, t0, 4);
      __builtin_memcpy(p + 2, t1, 4);
      Bd[(bcol[it] + 0) * 34 + brow[it]] = (_Float16)vb[it].x;
      Bd[(bcol[it] + 1) * 34 + brow[it]] = (_Float16)vb[it].y;
      Bd[(bcol[it] + 2) * 34 + brow[it]] = (_Float16)vb[it].z;
      Bd[(bcol[it] + 3) * 34 + brow[it]] = (_Float16)vb[it].w;
    }
  };

  // prologue: stage tile k=0 into buffer 0
  fetchT(0);
  stageT(0);

  int cur = 0;
  for (int k0 = 0; k0 < K; k0 += 32) {
    __syncthreads();                              // buf[cur] visible; old buf free
    bool hasNext = (k0 + 32 < K);                 // uniform
    if (hasNext) {
      fetchT(k0 + 32);                            // in flight during WMMAs below
      if (k0 + 64 < K) {                          // speculative prefetch 2 tiles ahead
        __builtin_prefetch(&Ab[(long)(mBase + arow[0]) * K + k0 + 64 + acol[0]], 0, 3);
        __builtin_prefetch(&Bb[(long)(k0 + 64 + brow[0]) * N + nBase + bcol[0]], 0, 3);
      }
    }

    // B fragment (shared by both m-subtiles)
    v16h bf;
    const _Float16* bp = &Bs[cur][((wc << 4) + ml) * 34];
#pragma unroll
    for (int v = 0; v < 8; ++v) {
      int kc = ((v < 4) ? (2 * v) : (16 + 2 * (v - 4))) + hi * 8;
      bf[2 * v] = bp[kc]; bf[2 * v + 1] = bp[kc + 1];
    }
    // A fragments for the two 16-row subtiles of this wave's band
    v16h af0, af1;
    const _Float16* ap0 = &As[cur][((wr << 5) + ml) * 34];
    const _Float16* ap1 = &As[cur][((wr << 5) + 16 + ml) * 34];
#pragma unroll
    for (int v = 0; v < 8; ++v) {
      int kb = ((v < 4) ? (2 * v) : (16 + 2 * (v - 4))) + hi * 8;
      af0[2 * v] = ap0[kb]; af0[2 * v + 1] = ap0[kb + 1];
      af1[2 * v] = ap1[kb]; af1[2 * v + 1] = ap1[kb + 1];
    }
    acc0 = __builtin_amdgcn_wmma_f32_16x16x32_f16(false, af0, false, bf, (short)0, acc0, false, false);
    acc1 = __builtin_amdgcn_wmma_f32_16x16x32_f16(false, af1, false, bf, (short)0, acc1, false, false);

    if (hasNext) stageT(cur ^ 1);                 // wait_loadcnt lands after WMMAs
    cur ^= 1;
  }

  float bn = bias ? bias[nc] : 0.0f;
#pragma unroll
  for (int q = 0; q < 8; ++q) {
    Cb[(long)(mw + q + 8 * hi) * N + nc]      = act_apply(acc0[q] + bn, act);
    Cb[(long)(mw + 16 + q + 8 * hi) * N + nc] = act_apply(acc1[q] + bn, act);
  }
}

// ---------------------------------------------------------------------------
// Channel einsum on WMMA: Y[g,d,t] = act(beta*Y + sum_c X[g,c,t]*W[c,d] + bias[d])
// One wave per (group, d-tile, t-tile). Fast path for C%32==0 && D%16==0
// (all expensive layers); guarded path uses clamped indices + selects
// (branch-free) for the tiny-C layers.
// ---------------------------------------------------------------------------
__global__ void chan_mm_wmma(const float* __restrict__ X, const float* __restrict__ W,
                             const float* __restrict__ bias, float* __restrict__ Y,
                             long G, int C, int D, int T, int beta, int act)
{
  int lane = threadIdx.x & 31;
  int wv   = threadIdx.x >> 5;
  long tile = (long)blockIdx.x * WPB + wv;
  int mt = (D + 15) >> 4, nt = T >> 4;
  long tpg = (long)mt * nt;
  long g = tile / tpg;
  if (g >= G) return;                             // wave-uniform
  long rr = tile - g * tpg;
  int tm = (int)(rr / nt), tn = (int)(rr % nt);
  int m0 = tm << 4, n0 = tn << 4;
  int ml = lane & 15, hi = lane >> 4;
  int ncol = n0 + ml;
  const float* Xg = X + g * (long)C * T;
  float*       Yg = Y + g * (long)D * T;
  int ma = m0 + ml;                               // output-channel row of A

  v8f acc = {0.f, 0.f, 0.f, 0.f, 0.f, 0.f, 0.f, 0.f};
  bool fast = ((C & 31) == 0) && ((D & 15) == 0);

  if (beta) {
#pragma unroll
    for (int q = 0; q < 8; ++q) {
      int m = m0 + q + 8 * hi;
      int mc = m < D ? m : (D - 1);               // clamped, always-safe load
      float vv = Yg[(long)mc * T + ncol];
      acc[q] = (m < D) ? vv : 0.f;
    }
  }

  if (fast) {
    for (int k0 = 0; k0 < C; k0 += 32) {
      float av[16], bv[16];
#pragma unroll
      for (int v = 0; v < 8; ++v) {
        int kb = k0 + ((v < 4) ? (2 * v) : (16 + 2 * (v - 4))) + hi * 8;
        av[2 * v]     = W[(long)kb * D + ma];
        av[2 * v + 1] = W[(long)(kb + 1) * D + ma];
        int kc = k0 + ((v < 4) ? (2 * v) : (16 + 2 * (v - 4))) + hi * 8;
        bv[2 * v]     = Xg[(long)kc * T + ncol];
        bv[2 * v + 1] = Xg[(long)(kc + 1) * T + ncol];
      }
      v16h af, bf;
#pragma unroll
      for (int j = 0; j < 16; ++j) { af[j] = (_Float16)av[j]; bf[j] = (_Float16)bv[j]; }
      acc = __builtin_amdgcn_wmma_f32_16x16x32_f16(false, af, false, bf, (short)0, acc, false, false);
    }
  } else {
    int mc = ma < D ? ma : (D - 1);
    for (int k0 = 0; k0 < C; k0 += 32) {
      float av[16], bv[16];
#pragma unroll
      for (int v = 0; v < 8; ++v) {
        int kb = k0 + ((v < 4) ? (2 * v) : (16 + 2 * (v - 4))) + hi * 8;
        int k1 = kb < C ? kb : (C - 1);
        int k2 = (kb + 1) < C ? (kb + 1) : (C - 1);
        float a0 = W[(long)k1 * D + mc];          // clamped loads, no branches
        float a1 = W[(long)k2 * D + mc];
        av[2 * v]     = (ma < D && kb     < C) ? a0 : 0.f;
        av[2 * v + 1] = (ma < D && kb + 1 < C) ? a1 : 0.f;
        int kc = k0 + ((v < 4) ? (2 * v) : (16 + 2 * (v - 4))) + hi * 8;
        int k3 = kc < C ? kc : (C - 1);
        int k4 = (kc + 1) < C ? (kc + 1) : (C - 1);
        float b0 = Xg[(long)k3 * T + ncol];
        float b1 = Xg[(long)k4 * T + ncol];
        bv[2 * v]     = (kc     < C) ? b0 : 0.f;
        bv[2 * v + 1] = (kc + 1 < C) ? b1 : 0.f;
      }
      v16h af, bf;
#pragma unroll
      for (int j = 0; j < 16; ++j) { af[j] = (_Float16)av[j]; bf[j] = (_Float16)bv[j]; }
      acc = __builtin_amdgcn_wmma_f32_16x16x32_f16(false, af, false, bf, (short)0, acc, false, false);
    }
  }

#pragma unroll
  for (int q = 0; q < 8; ++q) {
    int m = m0 + q + 8 * hi;
    if (m < D) {
      float vv = acc[q] + (bias ? bias[m] : 0.0f);
      Yg[(long)m * T + ncol] = act_apply(vv, act);
    }
  }
}

// ---------------------------------------------------------------------------
// Pointwise / helper kernels
// ---------------------------------------------------------------------------
__global__ void zero_k(float* __restrict__ p, long n) {
  long i = (long)blockIdx.x * blockDim.x + threadIdx.x;
  if (i < n) p[i] = 0.0f;
}
__global__ void copy_k(float* __restrict__ d, const float* __restrict__ s, long n) {
  long i = (long)blockIdx.x * blockDim.x + threadIdx.x;
  if (i < n) d[i] = s[i];
}
__global__ void add_k(float* __restrict__ d, const float* __restrict__ a,
                      const float* __restrict__ b, long n) {
  long i = (long)blockIdx.x * blockDim.x + threadIdx.x;
  if (i < n) d[i] = a[i] + b[i];
}
__global__ void edgew_k(const float* __restrict__ ea, const float* __restrict__ we,
                        float* __restrict__ w, int E) {
  int e = blockIdx.x * blockDim.x + threadIdx.x;
  if (e < E) w[e] = ea[3 * e] * we[0] + ea[3 * e + 1] * we[1] + ea[3 * e + 2] * we[2];
}
// agg[b,dst,c,t] += x[b,src,c,t] * w[e]   (native f32 global atomics)
__global__ void scatter_k(const float* __restrict__ x, const int* __restrict__ ei,
                          const float* __restrict__ w, float* __restrict__ agg,
                          int Bc, int V, int C, int T, int E) {
  long tid = (long)blockIdx.x * blockDim.x + threadIdx.x;
  long CT = (long)C * T;
  long total = (long)E * Bc * CT;
  if (tid >= total) return;
  int  e   = (int)(tid / ((long)Bc * CT));
  long rem = tid - (long)e * Bc * CT;
  int  b   = (int)(rem / CT);
  long ct  = rem - (long)b * CT;
  int s = ei[e], d = ei[E + e];
  float val = x[((long)b * V + s) * CT + ct] * w[e];
  __hip_atomic_fetch_add(&agg[((long)b * V + d) * CT + ct], val,
                         __ATOMIC_RELAXED, __HIP_MEMORY_SCOPE_AGENT);
}
// Fused GRU cell: h' = (1-z)*n + z*h, gi/gh are [G,192], h [G,64]
__global__ void gru_cell_k(const float* __restrict__ gi, const float* __restrict__ gh,
                           const float* __restrict__ hprev, float* __restrict__ hout, long G) {
  long i = (long)blockIdx.x * blockDim.x + threadIdx.x;
  if (i >= G * 64) return;
  long g = i >> 6; int j = (int)(i & 63);
  const float* gig = gi + g * 192;
  const float* ghg = gh + g * 192;
  float r = 1.0f / (1.0f + __expf(-(gig[j]       + ghg[j])));
  float z = 1.0f / (1.0f + __expf(-(gig[64 + j]  + ghg[64 + j])));
  float n = tanhf(gig[128 + j] + r * ghg[128 + j]);
  hout[i] = (1.0f - z) * n + z * hprev[i];
}
// h[g,c,t] -> xt[t,g,c]   (C=T=64)
__global__ void tr_fwd_k(const float* __restrict__ h, float* __restrict__ xt, long G) {
  long i = (long)blockIdx.x * blockDim.x + threadIdx.x;
  if (i >= G * 4096) return;
  long g = i / 4096; int rem = (int)(i - g * 4096);
  int c = rem >> 6, t = rem & 63;
  xt[((long)t * G + g) * 64 + c] = h[i];
}
// xt[t,g,c] -> h[g,c,t]
__global__ void tr_bwd_k(const float* __restrict__ xt, float* __restrict__ h, long G) {
  long i = (long)blockIdx.x * blockDim.x + threadIdx.x;
  if (i >= G * 4096) return;
  long g = i / 4096; int rem = (int)(i - g * 4096);
  int c = rem >> 6, t = rem & 63;
  h[i] = xt[((long)t * G + g) * 64 + c];
}

// ---------------------------------------------------------------------------
// Host orchestration
// ---------------------------------------------------------------------------
extern "C" void kernel_launch(void* const* d_in, const int* in_sizes, int n_in,
                              void* d_out, int out_size, void* d_ws, size_t ws_size,
                              hipStream_t stream) {
  (void)in_sizes; (void)n_in; (void)out_size; (void)ws_size;
  const int NFa[6] = {1, 8, 16, 32, 64, 128};
  const int Va[5]  = {2048, 1024, 512, 256, 128};
  const int Bn = 16, Tn = 64;

  // ---- input map (setup_inputs dict insertion order, depth-first) ----
  const float* x = (const float*)d_in[0];
  const float *cWs[4], *cWm[4], *cB[4], *cWe[4];
  for (int i = 0; i < 4; ++i) {
    cWs[i] = (const float*)d_in[1 + 4 * i]; cWm[i] = (const float*)d_in[2 + 4 * i];
    cB[i]  = (const float*)d_in[3 + 4 * i]; cWe[i] = (const float*)d_in[4 + 4 * i];
  }
  const float *dWs[4], *dWm[4], *dB[4], *dWe[4];
  for (int i = 0; i < 4; ++i) {
    dWs[i] = (const float*)d_in[17 + 4 * i]; dWm[i] = (const float*)d_in[18 + 4 * i];
    dB[i]  = (const float*)d_in[19 + 4 * i]; dWe[i] = (const float*)d_in[20 + 4 * i];
  }
  const float* fce1W = (const float*)d_in[33]; const float* fce1b = (const float*)d_in[34];
  const float* fce2W = (const float*)d_in[35]; const float* fce2b = (const float*)d_in[36];
  const float* fcd3W = (const float*)d_in[37]; const float* fcd3b = (const float*)d_in[38];
  const float* fcd4W = (const float*)d_in[39]; const float* fcd4b = (const float*)d_in[40];
  const float* embWi = (const float*)d_in[41]; const float* embWh = (const float*)d_in[42];
  const float* embbi = (const float*)d_in[43]; const float* embbh = (const float*)d_in[44];
  const float* domW1 = (const float*)d_in[45]; const float* domb1 = (const float*)d_in[46];
  const float* domW2 = (const float*)d_in[47]; const float* domb2 = (const float*)d_in[48];
  const float* odeW  = (const float*)d_in[49]; const float* odeb  = (const float*)d_in[50];
  const float* corWi = (const float*)d_in[51]; const float* corWh = (const float*)d_in[52];
  const float* corbi = (const float*)d_in[53]; const float* corbh = (const float*)d_in[54];
  const int*   ei[4]; for (int i = 0; i < 4; ++i) ei[i] = (const int*)d_in[55 + i];
  const float* ea[4]; for (int i = 0; i < 4; ++i) ea[i] = (const float*)d_in[60 + i];
  const float* Pd[4]; for (int i = 0; i < 4; ++i) Pd[i] = (const float*)d_in[65 + i];
  const float* Pu[4]; for (int i = 0; i < 4; ++i) Pu[i] = (const float*)d_in[69 + i];

  // ---- workspace layout (floats) ----
  float* wsf = (float*)d_ws;
  const long BIGN = 16777216L;       // 64 MB each
  float* big0 = wsf;
  float* big1 = wsf + BIGN;
  float* big2 = wsf + 2 * BIGN;      // agg / zseq scratch
  float* sb   = wsf + 3 * BIGN;
  float* state = sb;                 // 2048*64
  float* zDv   = sb + 131072;
  float* tmpv  = sb + 262144;
  float* fv    = sb + 393216;
  float* hpv   = sb + 524288;
  float* giv   = sb + 655360;        // 2048*192
  float* ghv   = sb + 1048576;       // 2048*192
  float* wed   = sb + 1441792;       // 8 * 16384 edge weights

  auto blocks1 = [](long n) { return (unsigned)((n + TPB - 1) / TPB); };
  auto gemm = [&](const float* A, const float* Bm, const float* bias, float* Cm,
                  int M, int N, int K, int nb, long sA, long sB, long sC, int beta, int act) {
    unsigned blk = (unsigned)((long)nb * (M >> 6) * (N >> 6));
    gemm_nn_wmma<<<dim3(blk), dim3(TPB), 0, stream>>>(A, Bm, bias, Cm, M, N, K, nb, sA, sB, sC, beta, act);
  };
  auto chan = [&](const float* X, const float* W, const float* bias, float* Y,
                  long G, int C, int D, int beta, int act) {
    long tiles = G * ((D + 15) >> 4) * (Tn >> 4);
    unsigned blk = (unsigned)((tiles + WPB - 1) / WPB);
    chan_mm_wmma<<<dim3(blk), dim3(TPB), 0, stream>>>(X, W, bias, Y, G, C, D, Tn, beta, act);
  };
  auto spatial = [&](const float* xin, float* agg, float* out, const int* eidx, const float* wE,
                     const float* Wself, const float* Wmsg, const float* bias,
                     int V, int Cin, int Cout, int E) {
    long na = (long)Bn * V * Cin * Tn;
    zero_k<<<blocks1(na), TPB, 0, stream>>>(agg, na);
    long tot = (long)E * Bn * Cin * Tn;
    scatter_k<<<blocks1(tot), TPB, 0, stream>>>(xin, eidx, wE, agg, Bn, V, Cin, Tn, E);
    chan(xin, Wself, nullptr, out, (long)Bn * V, Cin, Cout, 0, 0);
    chan(agg,  Wmsg,  bias,   out, (long)Bn * V, Cin, Cout, 1, 1);   // + bias, ELU
  };

  // ---- edge weights (4 conv levels + 4 deconv levels) ----
  for (int i = 0; i < 4; ++i) {
    int E = Va[i] * 8;
    edgew_k<<<blocks1(E), TPB, 0, stream>>>(ea[i], cWe[i], wed + i * 16384, E);
  }
  for (int i = 0; i < 4; ++i) {
    int E = Va[3 - i] * 8;
    edgew_k<<<blocks1(E), TPB, 0, stream>>>(ea[3 - i], dWe[i], wed + (4 + i) * 16384, E);
  }

  // ---- encoder: 4x (spatial block -> pooling GEMM) ----
  const float* hin = x;
  for (int i = 0; i < 4; ++i) {
    spatial(hin, big2, big0, ei[i], wed + i * 16384, cWs[i], cWm[i], cB[i],
            Va[i], NFa[i], NFa[i + 1], Va[i] * 8);
    int M = Va[i + 1], K = Va[i], N = NFa[i + 1] * Tn;
    gemm(Pd[i], big0, nullptr, big1, M, N, K, Bn, 0, (long)K * N, (long)M * N, 0, 0);
    hin = big1;
  }
  // fce1 (ELU) / fce2 (tanh)
  long G = (long)Bn * 128;  // 2048 group rows at level 4
  chan(big1, fce1W, fce1b, big0, G, 64, 128, 0, 1);
  chan(big0, fce2W, fce2b, big1, G, 128, 64, 0, 2);

  // ---- time modeling ----
  tr_fwd_k<<<blocks1(G * 4096), TPB, 0, stream>>>(big1, big0, G);  // xt[t,g,c]
  float* xt = big0;
  float* zt = big2;                                                // zseq[t,g,c]
  // embedding GRU (h0 = 0)
  zero_k<<<blocks1(G * 64), TPB, 0, stream>>>(state, G * 64);
  for (int t = 0; t < 64; ++t) {
    const float* xtt = xt + (long)t * G * 64;
    gemm(xtt,   embWi, embbi, giv, (int)G, 192, 64, 1, 0, 0, 0, 0, 0);
    gemm(state, embWh, embbh, ghv, (int)G, 192, 64, 1, 0, 0, 0, 0, 0);
    gru_cell_k<<<blocks1(G * 64), TPB, 0, stream>>>(giv, ghv, state, state, G);
  }
  // domain MLP: zD = tanh(elu(hT@W1+b1)@W2+b2)
  gemm(state, domW1, domb1, tmpv, (int)G, 64, 64, 1, 0, 0, 0, 0, 1);
  gemm(tmpv,  domW2, domb2, zDv,  (int)G, 64, 64, 1, 0, 0, 0, 0, 2);
  // ODE + correction scan: carry = xt[0]
  copy_k<<<blocks1(G * 64), TPB, 0, stream>>>(state, xt, G * 64);
  copy_k<<<blocks1(G * 64), TPB, 0, stream>>>(zt, xt, G * 64);     // zseq[0]
  for (int t = 1; t < 64; ++t) {
    // f = tanh([hprev, zD] @ W + b) via row-split accumulation
    gemm(state, odeW,            nullptr, fv, (int)G, 64, 64, 1, 0, 0, 0, 0, 0);
    gemm(zDv,   odeW + 64 * 64,  odeb,    fv, (int)G, 64, 64, 1, 0, 0, 0, 1, 2);
    add_k<<<blocks1(G * 64), TPB, 0, stream>>>(hpv, state, fv, G * 64);  // Euler step
    const float* xtt = xt + (long)t * G * 64;
    gemm(xtt, corWi, corbi, giv, (int)G, 192, 64, 1, 0, 0, 0, 0, 0);
    gemm(hpv, corWh, corbh, ghv, (int)G, 192, 64, 1, 0, 0, 0, 0, 0);
    gru_cell_k<<<blocks1(G * 64), TPB, 0, stream>>>(giv, ghv, hpv, state, G);
    copy_k<<<blocks1(G * 64), TPB, 0, stream>>>(zt + (long)t * G * 64, state, G * 64);
  }
  tr_bwd_k<<<blocks1(G * 4096), TPB, 0, stream>>>(zt, big1, G);    // back to [g,c,t]

  // ---- decoder ----
  chan(big1, fcd3W, fcd3b, big0, G, 64, 128, 0, 1);
  chan(big0, fcd4W, fcd4b, big1, G, 128, 64, 0, 1);
  for (int i = 0; i < 4; ++i) {
    int Vold = Va[4 - i], Vnew = Va[3 - i], Cc = NFa[4 - i];
    int M = Vnew, K = Vold, N = Cc * Tn;
    gemm(Pu[i], big1, nullptr, big0, M, N, K, Bn, 0, (long)K * N, (long)M * N, 0, 0);
    float* outp = (i < 3) ? big1 : (float*)d_out;
    spatial(big0, big2, outp, ei[3 - i], wed + (4 + i) * 16384, dWs[i], dWm[i], dB[i],
            Vnew, Cc, NFa[3 - i], Vnew * 8);
  }
}
